// TokenVisionTransformerMoE_22771916603560
// MI455X (gfx1250) — matrix operand order repaired and checked by reading, not compiled, not run on hardware
//
#include <hip/hip_runtime.h>
#include <hip/hip_bf16.h>
#include <math.h>

// ---------------------------------------------------------------------------
// Problem constants (from reference)
// ---------------------------------------------------------------------------
#define BATCH   32
#define IMG     224
#define PATCH   16
#define DMODEL  768
#define LAYERS  4
#define NHEADS  12
#define TASKS   2
#define HMLP    3072
#define NTOK    197            // 196 patches + cls
#define PTOK    196
#define HD      64             // head dim
#define TOKENS  (BATCH * NTOK)   // 6304
#define PTOKENS (BATCH * PTOK)   // 6272
#define LN_EPS  1e-6f
#define ATT_SCALE 0.125f       // 64^-0.5

// ---------------------------------------------------------------------------
// CDNA5 WMMA types
// ---------------------------------------------------------------------------
typedef __bf16 bf16_t;
typedef bf16_t v16bf __attribute__((ext_vector_type(16)));
typedef bf16_t v8bf  __attribute__((ext_vector_type(8)));
typedef float  v8f   __attribute__((ext_vector_type(8)));

union Frag16 { v16bf v; v8bf h[2]; };

__device__ __forceinline__ float gelu_exact(float x) {
    return 0.5f * x * (1.0f + erff(x * 0.70710678118654752f));
}

__device__ __forceinline__ float waveReduceSum(float v) {
    #pragma unroll
    for (int o = 16; o > 0; o >>= 1) v += __shfl_xor(v, o, 32);
    return v;
}
__device__ __forceinline__ float waveReduceMax(float v) {
    #pragma unroll
    for (int o = 16; o > 0; o >>= 1) v = fmaxf(v, __shfl_xor(v, o, 32));
    return v;
}

// ---------------------------------------------------------------------------
// Tiled bf16 WMMA GEMM:  C[M,Nout] = A[M,K] * W[Nout,K]^T   (+ epilogue)
//   block tile 128x128, 8 waves (wave32) arranged 2(M) x 4(N);
//   each wave owns a 64x32 tile = 4x2 fragments -> 8x v_wmma_f32_16x16x32_bf16
//   per K=32 chunk from 4 A-frags + 2 B-frags (6x 32B LDS reads / 8 WMMAs).
//   Global->reg loads are issued BEFORE the barrier so they overlap the
//   previous chunk's WMMAs; cvt+ds_store co-executes with the XDL pipe.
// Epilogue: 0 = +bias ; 1 = gelu(x+bias) ; 2 = x + bias + res
// ---------------------------------------------------------------------------
#define BM 128
#define BN 128
#define BK 32
#define LDA 40   // BK + 8 pad (bf16 elems) -> 80B row stride, 16B aligned
#define LDB 40

template <int EPI>
__global__ __launch_bounds__(256)
void gemm_bf16_wmma(const float* __restrict__ A,
                    const float* __restrict__ W,
                    const float* __restrict__ bias,
                    const float* __restrict__ res,
                    float* __restrict__ C,
                    int M, int K, int Nout)
{
    __shared__ bf16_t Als[BM * LDA];
    __shared__ bf16_t Bls[BN * LDB];

    const int tid  = threadIdx.x;
    const int wave = tid >> 5;
    const int lane = tid & 31;
    const int rowBase = (wave >> 2) * 64;   // 2 wave-rows
    const int colBase = (wave & 3) * 32;    // 4 wave-cols
    const int half = lane >> 4;             // 0 or 1
    const int m16  = lane & 15;

    const int bm = blockIdx.x * BM;
    const int bn = blockIdx.y * BN;

    // cooperative-fill coordinates (fixed per thread): 8 float4 per thread
    const int fr[4] = { tid >> 3, (tid + 256) >> 3, (tid + 512) >> 3, (tid + 768) >> 3 };
    const int fcv   = (tid & 7) * 4;

    v8f acc[4][2];
    #pragma unroll
    for (int i = 0; i < 4; ++i)
        #pragma unroll
        for (int j = 0; j < 2; ++j)
            #pragma unroll
            for (int r = 0; r < 8; ++r)
                acc[i][j][r] = 0.0f;

    for (int k0 = 0; k0 < K; k0 += BK) {
        // ---- phase 1: issue all global loads into registers (overlaps WMMAs)
        float4 av[4], bv[4];
        #pragma unroll
        for (int it = 0; it < 4; ++it) {
            int grow = bm + fr[it];
            av[it] = make_float4(0.f, 0.f, 0.f, 0.f);
            if (grow < M) {
                const float* ap = A + (size_t)grow * K + k0 + fcv;
                av[it] = *(const float4*)ap;
                if (k0 + BK < K) __builtin_prefetch(ap + BK, 0, 1);  // global_prefetch_b8
            }
        }
        #pragma unroll
        for (int it = 0; it < 4; ++it) {
            const float* wp = W + (size_t)(bn + fr[it]) * K + k0 + fcv;
            bv[it] = *(const float4*)wp;
            if (k0 + BK < K) __builtin_prefetch(wp + BK, 0, 1);
        }

        __syncthreads();   // previous chunk's fragment loads complete

        // ---- phase 2: convert fp32 -> bf16 and store to LDS
        #pragma unroll
        for (int it = 0; it < 4; ++it) {
            bf16_t* dst = &Als[fr[it] * LDA + fcv];
            dst[0] = (bf16_t)av[it].x; dst[1] = (bf16_t)av[it].y;
            dst[2] = (bf16_t)av[it].z; dst[3] = (bf16_t)av[it].w;
        }
        #pragma unroll
        for (int it = 0; it < 4; ++it) {
            bf16_t* dst = &Bls[fr[it] * LDB + fcv];
            dst[0] = (bf16_t)bv[it].x; dst[1] = (bf16_t)bv[it].y;
            dst[2] = (bf16_t)bv[it].z; dst[3] = (bf16_t)bv[it].w;
        }

        __syncthreads();

        // ---- phase 3: fragment loads per ISA VGPR striping + 8 WMMAs
        Frag16 fa[4], fb[2];
        const int kOffA = half * 8;   // lanes 0-15: K 0-7/16-23 ; 16-31: K 8-15/24-31
        #pragma unroll
        for (int i = 0; i < 4; ++i) {
            const bf16_t* ap = &Als[(rowBase + i * 16 + m16) * LDA + kOffA];
            fa[i].h[0] = *(const v8bf*)(ap);
            fa[i].h[1] = *(const v8bf*)(ap + 16);
        }
        #pragma unroll
        for (int j = 0; j < 2; ++j) {
            const bf16_t* bp = &Bls[(colBase + j * 16 + m16) * LDB + half * 8];
            fb[j].h[0] = *(const v8bf*)(bp);
            fb[j].h[1] = *(const v8bf*)(bp + 16);
        }
        #pragma unroll
        for (int i = 0; i < 4; ++i)
            #pragma unroll
            for (int j = 0; j < 2; ++j)
                acc[i][j] = __builtin_amdgcn_wmma_f32_16x16x32_bf16(
                    false, fa[i].v, false, fb[j].v,
                    (short)0, acc[i][j], false, false);
    }

    // ---- epilogue & store (C layout: VGPR r, lane<16 -> M rows 0..7, lane>=16 -> 8..15)
    #pragma unroll
    for (int i = 0; i < 4; ++i) {
        #pragma unroll
        for (int j = 0; j < 2; ++j) {
            int col = bn + colBase + j * 16 + m16;
            float bvv = bias ? bias[col] : 0.0f;
            #pragma unroll
            for (int r = 0; r < 8; ++r) {
                int row = bm + rowBase + i * 16 + half * 8 + r;
                if (row < M) {
                    float v = acc[i][j][r] + bvv;
                    if (EPI == 1) v = gelu_exact(v);
                    if (EPI == 2) v += res[(size_t)row * Nout + col];
                    C[(size_t)row * Nout + col] = v;
                }
            }
        }
    }
}

// ---------------------------------------------------------------------------
// im2col for the stride-16 patch conv: x[B,3,224,224] -> Acol[6272, 768]
//   col c = ch*256 + iy*16 + ix  (matches patch_w [D][3*16*16] row-major)
// ---------------------------------------------------------------------------
__global__ __launch_bounds__(256)
void im2col_kernel(const float* __restrict__ x, float* __restrict__ Acol)
{
    int idx = blockIdx.x * 256 + threadIdx.x;
    if (idx >= PTOKENS * DMODEL) return;
    int tok = idx / DMODEL;
    int c   = idx - tok * DMODEL;
    int b   = tok / PTOK;
    int p   = tok - b * PTOK;
    int py  = p / 14, px = p - py * 14;
    int ch  = c >> 8;
    int rem = c & 255;
    int iy  = rem >> 4, ix = rem & 15;
    Acol[idx] = x[(((size_t)b * 3 + ch) * IMG + py * PATCH + iy) * IMG + px * PATCH + ix];
}

// ---------------------------------------------------------------------------
// Assemble tokens: z[b][0] = cls + pos[0]; z[b][1+p] = patch_out[b*196+p] + pos[1+p]
// ---------------------------------------------------------------------------
__global__ __launch_bounds__(256)
void assemble_kernel(const float* __restrict__ patch_out,
                     const float* __restrict__ cls_tok,
                     const float* __restrict__ pos,
                     float* __restrict__ z)
{
    int idx = blockIdx.x * 256 + threadIdx.x;
    if (idx >= TOKENS * DMODEL) return;
    int tok = idx / DMODEL;
    int d   = idx - tok * DMODEL;
    int b   = tok / NTOK;
    int n   = tok - b * NTOK;
    float v;
    if (n == 0) v = cls_tok[d];
    else        v = patch_out[((size_t)b * PTOK + (n - 1)) * DMODEL + d];
    z[idx] = v + pos[(size_t)n * DMODEL + d];
}

// ---------------------------------------------------------------------------
// LayerNorm over D=768, one block (256 thr, 8 waves) per token
// ---------------------------------------------------------------------------
__global__ __launch_bounds__(256)
void ln_kernel(const float* __restrict__ x, const float* __restrict__ w,
               const float* __restrict__ b, float* __restrict__ out)
{
    __shared__ float s1[8], s2[8];
    const int tok  = blockIdx.x;
    const int tid  = threadIdx.x;
    const int wave = tid >> 5, lane = tid & 31;
    const float* xr = x + (size_t)tok * DMODEL;

    float v0 = xr[tid], v1 = xr[tid + 256], v2 = xr[tid + 512];
    float s  = v0 + v1 + v2;
    float sq = v0 * v0 + v1 * v1 + v2 * v2;
    s = waveReduceSum(s); sq = waveReduceSum(sq);
    if (lane == 0) { s1[wave] = s; s2[wave] = sq; }
    __syncthreads();
    float ts = 0.f, tq = 0.f;
    #pragma unroll
    for (int i = 0; i < 8; ++i) { ts += s1[i]; tq += s2[i]; }
    float m    = ts * (1.0f / DMODEL);
    float var  = tq * (1.0f / DMODEL) - m * m;
    float rstd = rsqrtf(var + LN_EPS);
    float* orow = out + (size_t)tok * DMODEL;
    orow[tid]       = (v0 - m) * rstd * w[tid]       + b[tid];
    orow[tid + 256] = (v1 - m) * rstd * w[tid + 256] + b[tid + 256];
    orow[tid + 512] = (v2 - m) * rstd * w[tid + 512] + b[tid + 512];
}

// ---------------------------------------------------------------------------
// Attention: one block per (b, head), 8 waves, each wave owns q rows
//   qkv layout [token, 3*768]; q@0, k@768, v@1536, head slice h*64.
//   (~1% of total FLOPs -> VALU path; GEMMs carry the WMMA throughput)
// ---------------------------------------------------------------------------
__global__ __launch_bounds__(256)
void attn_kernel(const float* __restrict__ qkv, float* __restrict__ o)
{
    __shared__ float qs[8][HD];
    __shared__ float ps[8][NTOK + 3];

    const int bh = blockIdx.x;
    const int b  = bh / NHEADS;
    const int h  = bh - b * NHEADS;
    const int wave = threadIdx.x >> 5, lane = threadIdx.x & 31;
    const float* base = qkv + (size_t)b * NTOK * (3 * DMODEL);

    for (int qi = wave; qi < NTOK; qi += 8) {
        const float* qrow = base + (size_t)qi * (3 * DMODEL) + h * HD;
        qs[wave][lane]      = qrow[lane];
        qs[wave][lane + 32] = qrow[lane + 32];
        asm volatile("s_wait_dscnt 0" ::: "memory");   // intra-wave LDS RAW

        // scores + running max (lane-parallel over keys)
        float mx = -3.0e38f;
        for (int j = lane; j < NTOK; j += 32) {
            const float* krow = base + (size_t)j * (3 * DMODEL) + DMODEL + h * HD;
            float s = 0.f;
            #pragma unroll
            for (int d = 0; d < HD; ++d) s += qs[wave][d] * krow[d];
            s *= ATT_SCALE;
            ps[wave][j] = s;
            mx = fmaxf(mx, s);
        }
        mx = waveReduceMax(mx);
        float sum = 0.f;
        for (int j = lane; j < NTOK; j += 32) {
            float e = __expf(ps[wave][j] - mx);
            ps[wave][j] = e;
            sum += e;
        }
        sum = waveReduceSum(sum);
        float inv = 1.0f / sum;
        asm volatile("s_wait_dscnt 0" ::: "memory");

        // o[d] = sum_j p_j * V[j][d]
        float* orow = o + ((size_t)b * NTOK + qi) * DMODEL + h * HD;
        #pragma unroll
        for (int dd = 0; dd < 2; ++dd) {
            int d = lane + dd * 32;
            float acc = 0.f;
            for (int j = 0; j < NTOK; ++j)
                acc += ps[wave][j] * base[(size_t)j * (3 * DMODEL) + 2 * DMODEL + h * HD + d];
            orow[d] = acc * inv;
        }
    }
}

// ---------------------------------------------------------------------------
// Router: probs[t][tok] = softmax(h . router_w[t])[1]   (128 thr / token)
// ---------------------------------------------------------------------------
__global__ __launch_bounds__(128)
void router_kernel(const float* __restrict__ h, const float* __restrict__ rw,
                   float* __restrict__ probs)
{
    __shared__ float sm[4][4];
    const int tok = blockIdx.x;
    const int tid = threadIdx.x;
    const int wave = tid >> 5, lane = tid & 31;
    float a00 = 0.f, a01 = 0.f, a10 = 0.f, a11 = 0.f;
    const float* hr = h + (size_t)tok * DMODEL;
    for (int d = tid; d < DMODEL; d += 128) {
        float hv = hr[d];
        a00 += hv * rw[(size_t)d * 2];
        a01 += hv * rw[(size_t)d * 2 + 1];
        a10 += hv * rw[((size_t)DMODEL + d) * 2];
        a11 += hv * rw[((size_t)DMODEL + d) * 2 + 1];
    }
    a00 = waveReduceSum(a00); a01 = waveReduceSum(a01);
    a10 = waveReduceSum(a10); a11 = waveReduceSum(a11);
    if (lane == 0) { sm[wave][0] = a00; sm[wave][1] = a01; sm[wave][2] = a10; sm[wave][3] = a11; }
    __syncthreads();
    if (tid == 0) {
        float l00 = 0.f, l01 = 0.f, l10 = 0.f, l11 = 0.f;
        #pragma unroll
        for (int i = 0; i < 4; ++i) { l00 += sm[i][0]; l01 += sm[i][1]; l10 += sm[i][2]; l11 += sm[i][3]; }
        probs[tok]          = 1.0f / (1.0f + __expf(l00 - l01));
        probs[TOKENS + tok] = 1.0f / (1.0f + __expf(l10 - l11));
    }
}

// ---------------------------------------------------------------------------
// MoE masked aggregation + final LayerNorm (256 thr / token)
//   valid_t == shared_t (since agg_mask = any_t shared); out = mask? agg : task0
// ---------------------------------------------------------------------------
__global__ __launch_bounds__(256)
void agg_lnf_kernel(const float* __restrict__ t0, const float* __restrict__ t1,
                    const float* __restrict__ probs,
                    const float* __restrict__ lw, const float* __restrict__ lb,
                    float* __restrict__ out)
{
    __shared__ float s1[8], s2[8];
    const int tok  = blockIdx.x;
    const int tid  = threadIdx.x;
    const int wave = tid >> 5, lane = tid & 31;

    bool sh0 = probs[tok] > 0.5f;
    bool sh1 = probs[TOKENS + tok] > 0.5f;
    bool mask = sh0 | sh1;
    float v0 = sh0 ? 1.f : 0.f;
    float v1 = sh1 ? 1.f : 0.f;
    float invc = 1.0f / (v0 + v1 + LN_EPS);

    const float* r0 = t0 + (size_t)tok * DMODEL;
    const float* r1 = t1 + (size_t)tok * DMODEL;
    float val[3];
    float s = 0.f, sq = 0.f;
    #pragma unroll
    for (int e = 0; e < 3; ++e) {
        int d = tid + e * 256;
        float a = r0[d], bb = r1[d];
        float agg = (a * v0 + bb * v1) * invc;
        float v = mask ? agg : a;
        val[e] = v;
        s += v; sq += v * v;
    }
    s = waveReduceSum(s); sq = waveReduceSum(sq);
    if (lane == 0) { s1[wave] = s; s2[wave] = sq; }
    __syncthreads();
    float ts = 0.f, tq = 0.f;
    #pragma unroll
    for (int i = 0; i < 8; ++i) { ts += s1[i]; tq += s2[i]; }
    float m    = ts * (1.0f / DMODEL);
    float var  = tq * (1.0f / DMODEL) - m * m;
    float rstd = rsqrtf(var + LN_EPS);
    float* orow = out + (size_t)tok * DMODEL;
    #pragma unroll
    for (int e = 0; e < 3; ++e) {
        int d = tid + e * 256;
        orow[d] = (val[e] - m) * rstd * lw[d] + lb[d];
    }
}

// ---------------------------------------------------------------------------
// Host orchestration
// ---------------------------------------------------------------------------
extern "C" void kernel_launch(void* const* d_in, const int* in_sizes, int n_in,
                              void* d_out, int out_size, void* d_ws, size_t ws_size,
                              hipStream_t stream)
{
    (void)in_sizes; (void)n_in; (void)out_size; (void)ws_size;

    const float* x        = (const float*)d_in[0];
    const float* patch_w  = (const float*)d_in[1];   // [768, 768] (D, 3*16*16)
    const float* patch_b  = (const float*)d_in[2];
    const float* cls_tok  = (const float*)d_in[3];
    const float* pos      = (const float*)d_in[4];
    const float* ln1_w    = (const float*)d_in[5];
    const float* ln1_b    = (const float*)d_in[6];
    const float* qkv_w    = (const float*)d_in[7];   // [L, 2304, 768]
    const float* qkv_b    = (const float*)d_in[8];
    const float* proj_w   = (const float*)d_in[9];   // [L, 768, 768]
    const float* proj_b   = (const float*)d_in[10];
    const float* ln2_w    = (const float*)d_in[11];
    const float* ln2_b    = (const float*)d_in[12];
    const float* fc1_w    = (const float*)d_in[13];  // [L, 3072, 768]
    const float* fc1_b    = (const float*)d_in[14];
    const float* fc2_w    = (const float*)d_in[15];  // [L, 768, 3072]
    const float* fc2_b    = (const float*)d_in[16];
    const float* ln3_w    = (const float*)d_in[17];
    const float* ln3_b    = (const float*)d_in[18];
    const float* router_w = (const float*)d_in[19];  // [T, 768, 2]
    const float* efc1_w   = (const float*)d_in[20];  // [T, 3072, 768]
    const float* efc1_b   = (const float*)d_in[21];
    const float* efc2_w   = (const float*)d_in[22];  // [T, 768, 3072]
    const float* efc2_b   = (const float*)d_in[23];
    const float* lnf_w    = (const float*)d_in[24];
    const float* lnf_b    = (const float*)d_in[25];
    float* out = (float*)d_out;

    // workspace layout (floats)
    float* ws   = (float*)d_ws;
    float* zb   = ws;                                  // [6304, 768]
    float* hb   = zb   + (size_t)TOKENS * DMODEL;      // [6304, 768]
    float* qkvb = hb   + (size_t)TOKENS * DMODEL;      // [6304, 2304]
    float* ob   = qkvb + (size_t)TOKENS * 3 * DMODEL;  // [6304, 768]
    float* mb   = ob   + (size_t)TOKENS * DMODEL;      // [6304, 3072] (also im2col)
    float* t0b  = mb   + (size_t)TOKENS * HMLP;        // [6304, 768]
    float* t1b  = t0b  + (size_t)TOKENS * DMODEL;      // [6304, 768]
    float* prb  = t1b  + (size_t)TOKENS * DMODEL;      // [2, 6304]

    dim3 blk(256);

    // ---- patch embedding: im2col -> WMMA GEMM (+bias) -> assemble tokens
    im2col_kernel<<<(PTOKENS * DMODEL + 255) / 256, blk, 0, stream>>>(x, mb);
    {
        dim3 grid((PTOKENS + BM - 1) / BM, DMODEL / BN);
        gemm_bf16_wmma<0><<<grid, blk, 0, stream>>>(mb, patch_w, patch_b, nullptr,
                                                    hb, PTOKENS, DMODEL, DMODEL);
    }
    assemble_kernel<<<(TOKENS * DMODEL + 255) / 256, blk, 0, stream>>>(hb, cls_tok, pos, zb);

    // ---- transformer blocks
    for (int i = 0; i < LAYERS; ++i) {
        ln_kernel<<<TOKENS, blk, 0, stream>>>(zb, ln1_w + i * DMODEL, ln1_b + i * DMODEL, hb);
        {
            dim3 grid((TOKENS + BM - 1) / BM, (3 * DMODEL) / BN);
            gemm_bf16_wmma<0><<<grid, blk, 0, stream>>>(
                hb, qkv_w + (size_t)i * 3 * DMODEL * DMODEL, qkv_b + (size_t)i * 3 * DMODEL,
                nullptr, qkvb, TOKENS, DMODEL, 3 * DMODEL);
        }
        attn_kernel<<<BATCH * NHEADS, blk, 0, stream>>>(qkvb, ob);
        {
            dim3 grid((TOKENS + BM - 1) / BM, DMODEL / BN);
            gemm_bf16_wmma<2><<<grid, blk, 0, stream>>>(
                ob, proj_w + (size_t)i * DMODEL * DMODEL, proj_b + (size_t)i * DMODEL,
                zb, zb, TOKENS, DMODEL, DMODEL);
        }
        ln_kernel<<<TOKENS, blk, 0, stream>>>(zb, ln2_w + i * DMODEL, ln2_b + i * DMODEL, hb);
        {
            dim3 grid((TOKENS + BM - 1) / BM, HMLP / BN);
            gemm_bf16_wmma<1><<<grid, blk, 0, stream>>>(
                hb, fc1_w + (size_t)i * HMLP * DMODEL, fc1_b + (size_t)i * HMLP,
                nullptr, mb, TOKENS, DMODEL, HMLP);
        }
        {
            dim3 grid((TOKENS + BM - 1) / BM, DMODEL / BN);
            gemm_bf16_wmma<2><<<grid, blk, 0, stream>>>(
                mb, fc2_w + (size_t)i * DMODEL * HMLP, fc2_b + (size_t)i * DMODEL,
                zb, zb, TOKENS, HMLP, DMODEL);
        }
    }

    // ---- MoE tail
    ln_kernel<<<TOKENS, blk, 0, stream>>>(zb, ln3_w, ln3_b, hb);
    router_kernel<<<TOKENS, dim3(128), 0, stream>>>(hb, router_w, prb);
    for (int t = 0; t < TASKS; ++t) {
        {
            dim3 grid((TOKENS + BM - 1) / BM, HMLP / BN);
            gemm_bf16_wmma<1><<<grid, blk, 0, stream>>>(
                hb, efc1_w + (size_t)t * HMLP * DMODEL, efc1_b + (size_t)t * HMLP,
                nullptr, mb, TOKENS, DMODEL, HMLP);
        }
        {
            dim3 grid((TOKENS + BM - 1) / BM, DMODEL / BN);
            gemm_bf16_wmma<2><<<grid, blk, 0, stream>>>(
                mb, efc2_w + (size_t)t * DMODEL * HMLP, efc2_b + (size_t)t * DMODEL,
                zb, (t == 0 ? t0b : t1b), TOKENS, HMLP, DMODEL);
        }
    }
    agg_lnf_kernel<<<TOKENS, blk, 0, stream>>>(t0b, t1b, prb, lnf_w, lnf_b, out);
}
